// Encoder_45037027066538
// MI455X (gfx1250) — matrix operand order repaired and checked by reading, not compiled, not run on hardware
//
#include <hip/hip_runtime.h>
#include <hip/hip_bf16.h>
#include <cstdint>
#include <cstddef>

// ---------------------------------------------------------------------------
// Types / WMMA helpers (CDNA5 / gfx1250, wave32)
// ---------------------------------------------------------------------------
typedef __attribute__((ext_vector_type(16))) _Float16 v16h;
typedef __attribute__((ext_vector_type(8)))  _Float16 v8h;
typedef __attribute__((ext_vector_type(8)))  float    v8f;

#define DEVINL __device__ __forceinline__

DEVINL v8f wmma_f16(v16h a, v16h b, v8f c) {
  // D = A(16x32 f16) * B(32x16 f16) + C(16x16 f32)
  return __builtin_amdgcn_wmma_f32_16x16x32_f16(
      /*neg_a=*/false, a, /*neg_b=*/false, b,
      /*c_mod=*/(short)0, c, /*reuse_a=*/false, /*reuse_b=*/false);
}

// A fragment: 16x32 f16 tile, row-major source with row stride lda (f16 elems).
// ISA layout: lanes 0-15 -> M=lane, K in {k0+0..7, k0+16..23};
//             lanes 16-31 -> M=lane-16, K in {k0+8..15, k0+24..31}.
DEVINL v16h a_frag(const _Float16* A, int lda, int k0, int lane) {
  const int m   = lane & 15;
  const int off = (lane & 16) ? 8 : 0;
  const _Float16* p = A + m * lda + k0 + off;
  v8h lo = *(const v8h*)(p);        // K = k0+off   .. k0+off+7
  v8h hi = *(const v8h*)(p + 16);   // K = k0+off+16.. k0+off+23
  v16h a;
#pragma unroll
  for (int j = 0; j < 8; ++j) { a[j] = lo[j]; a[j + 8] = hi[j]; }
  return a;
}

// B fragment: 32x16 f16 tile from a pre-transposed weight Wt[N][Kpad] (f16).
// ISA layout: lanes 0-15 -> N=n0+lane, K in {k0+0..7, k0+16..23};
//             lanes 16-31 -> N=n0+lane-16, K in {k0+8..15, k0+24..31}.
DEVINL v16h b_frag(const _Float16* Wt, int kpad, int n0, int k0, int lane) {
  const int nn = n0 + (lane & 15);
  const int kk = k0 + ((lane & 16) ? 8 : 0);
  const _Float16* p = Wt + (size_t)nn * kpad + kk;
  v8h lo = *(const v8h*)(p);
  v8h hi = *(const v8h*)(p + 16);
  v16h b;
#pragma unroll
  for (int j = 0; j < 8; ++j) { b[j] = lo[j]; b[j + 8] = hi[j]; }
  return b;
}

// One wave computes two adjacent 16x16 output tiles (shared A fragment).
template <int KTOT>
DEVINL void gemm2(const _Float16* A, int lda, const _Float16* Wt, int kpad,
                  int n0a, int n0b, int lane, v8f& c0, v8f& c1) {
  for (int k0 = 0; k0 < KTOT; k0 += 32) {
    v16h a  = a_frag(A, lda, k0, lane);
    v16h b0 = b_frag(Wt, kpad, n0a, k0, lane);
    v16h b1 = b_frag(Wt, kpad, n0b, k0, lane);
    c0 = wmma_f16(a, b0, c0);
    c1 = wmma_f16(a, b1, c1);
  }
}

// C/D layout: lane 0-15 -> N=n0+lane, VGPR r -> M=r; lane 16-31 -> M=r+8.
DEVINL void store_c_f32(float* C, int ldc, int n0, int lane, v8f c) {
  const int col = n0 + (lane & 15);
  const int mb  = (lane & 16) ? 8 : 0;
#pragma unroll
  for (int r = 0; r < 8; ++r) C[(mb + r) * ldc + col] = c[r];
}

// ---------------------------------------------------------------------------
// Problem constants
// ---------------------------------------------------------------------------
#define NB    64
#define TT    128
#define JJ    28
#define WW    16
#define EMB   128
#define CIN   196
#define CINP  224          // CIN padded to a multiple of 32 (zero-filled)
#define TP    144          // T + W (edge-padded)
#define NWIN  (NB * TT)    // 8192 windows
#define FF    512

// ---------------------------------------------------------------------------
// Prep: build padded f16 activation tensor Xpad[B][144][224]
// ---------------------------------------------------------------------------
__global__ void prep_x(const float* __restrict__ enc,   // [B][112][T]
                       const float* __restrict__ off,   // [B][28][3]
                       _Float16* __restrict__ xp) {     // [B][144][224]
  int idx = blockIdx.x * blockDim.x + threadIdx.x;
  if (idx >= NB * TP * CINP) return;
  int c  = idx % CINP;
  int tp = (idx / CINP) % TP;
  int b  = idx / (CINP * TP);
  float v = 0.f;
  if (c < 112) {
    int t = tp - 8; t = t < 0 ? 0 : (t > 127 ? 127 : t);   // edge pad
    v = enc[((size_t)b * 112 + c) * TT + t];
  } else if (c < CIN) {
    v = off[(size_t)b * 84 + (c - 112)];                    // broadcast over t
  }
  xp[idx] = (_Float16)v;
}

// Convert weight W[K][N] (f32) -> Wt[N][Kpad] (f16, transposed, zero K-pad)
__global__ void prep_w(const float* __restrict__ W, _Float16* __restrict__ Wt,
                       int K, int N, int Kpad) {
  int idx = blockIdx.x * blockDim.x + threadIdx.x;
  if (idx >= N * Kpad) return;
  int kp = idx % Kpad;
  int n  = idx / Kpad;
  Wt[idx] = (_Float16)((kp < K) ? W[(size_t)kp * N + n] : 0.f);
}

// ---------------------------------------------------------------------------
// Fused transformer block: one workgroup (4 wave32) per 16-token window.
// MODE 0: block1 (A from Xpad gather, no residual, f32 out -> x1)
// MODE 1: block2 (A from x1 f32, residual, f16 out -> x2 = x1 + block(x1))
// ---------------------------------------------------------------------------
template <int CK, int MODE>
__global__ void __launch_bounds__(128)
block_kernel(const _Float16* __restrict__ Xpad,   // MODE 0 source
             const float*    __restrict__ Xin,    // MODE 1 source (x1 f32)
             const _Float16* __restrict__ wqT, const _Float16* __restrict__ wkT,
             const _Float16* __restrict__ wvT, const _Float16* __restrict__ woT,
             const float* __restrict__ bo,
             const _Float16* __restrict__ ff1T, const float* __restrict__ bf1,
             const _Float16* __restrict__ ff2T, const float* __restrict__ bf2,
             const float* __restrict__ g1, const float* __restrict__ b1v,
             const float* __restrict__ g2, const float* __restrict__ b2v,
             float* __restrict__ outF32,          // MODE 0
             _Float16* __restrict__ outF16) {     // MODE 1
  __shared__ _Float16 sA[16 * CK];                      // GEMM-A (window, f16)
  __shared__ float    sRes[(MODE == 1) ? 16 * EMB : 16];// residual (block2)
  __shared__ union QF {                                 // QKV f32  <->  FF1 f16
    struct { float q[16 * EMB]; float k[16 * EMB]; float v[16 * EMB]; } qkv;
    _Float16 f[16 * FF];
  } su;
  __shared__ _Float16 sO[16 * EMB];    // attention out (f16, GEMM-A for wo)
  __shared__ float    sT[16 * EMB];    // pre-LN scratch
  __shared__ float    sH[16 * EMB];    // h (f32, residual into FF)
  __shared__ _Float16 sH16[16 * EMB];  // h (f16, GEMM-A for ff1)

  const int tid  = threadIdx.x;
  const int wave = tid >> 5;
  const int lane = tid & 31;
  const int win  = blockIdx.x;

  // ---- 1. stage window A matrix into LDS ----
  if (MODE == 0) {
    const int b = win >> 7, t = win & 127;
    const _Float16* src = Xpad + ((size_t)b * TP + t) * CINP;  // rows t..t+15
    for (int i = tid; i < 16 * CINP; i += 128) sA[i] = src[i];
  } else {
    const float* src = Xin + (size_t)win * (16 * EMB);
    for (int i = tid; i < 16 * EMB; i += 128) {
      float v = src[i];
      sA[i] = (_Float16)v;
      if constexpr (MODE == 1) sRes[i] = v;
    }
  }
  __syncthreads();

  // ---- 2. QKV GEMMs: (16 x CK) @ (CK x 128), each wave owns 2 N-tiles ----
  {
    const int base = wave * 32;
    v8f q0 = {}, q1 = {}, k0 = {}, k1 = {}, v0 = {}, v1 = {};
    gemm2<CK>(sA, CK, wqT, CK, base, base + 16, lane, q0, q1);
    gemm2<CK>(sA, CK, wkT, CK, base, base + 16, lane, k0, k1);
    gemm2<CK>(sA, CK, wvT, CK, base, base + 16, lane, v0, v1);
    store_c_f32(su.qkv.q, EMB, base,      lane, q0);
    store_c_f32(su.qkv.q, EMB, base + 16, lane, q1);
    store_c_f32(su.qkv.k, EMB, base,      lane, k0);
    store_c_f32(su.qkv.k, EMB, base + 16, lane, k1);
    store_c_f32(su.qkv.v, EMB, base,      lane, v0);
    store_c_f32(su.qkv.v, EMB, base + 16, lane, v1);
  }
  __syncthreads();

  // ---- 3. attention, f32: 64 heads x head-dim 2, softmax over 16 keys ----
  {
    const float scale = 0.70710678118654752f;  // 1/sqrt(2)
    for (int item = tid; item < 64 * 16; item += 128) {
      const int h  = item >> 4;
      const int qi = item & 15;
      const float q0 = su.qkv.q[qi * EMB + 2 * h];
      const float q1 = su.qkv.q[qi * EMB + 2 * h + 1];
      float s[16];
      float mx = -1e30f;
#pragma unroll
      for (int j = 0; j < 16; ++j) {
        float sv = (q0 * su.qkv.k[j * EMB + 2 * h] +
                    q1 * su.qkv.k[j * EMB + 2 * h + 1]) * scale;
        s[j] = sv;
        mx = fmaxf(mx, sv);
      }
      float sum = 0.f, o0 = 0.f, o1 = 0.f;
#pragma unroll
      for (int j = 0; j < 16; ++j) {
        float e = __expf(s[j] - mx);
        sum += e;
        o0 += e * su.qkv.v[j * EMB + 2 * h];
        o1 += e * su.qkv.v[j * EMB + 2 * h + 1];
      }
      const float inv = 1.f / sum;
      sO[qi * EMB + 2 * h]     = (_Float16)(o0 * inv);
      sO[qi * EMB + 2 * h + 1] = (_Float16)(o1 * inv);
    }
  }
  __syncthreads();

  // ---- 4. o @ wo + bo (+ residual) -> sT ----
  {
    const int base = wave * 32;
    v8f c0 = {}, c1 = {};
    gemm2<EMB>(sO, EMB, woT, EMB, base, base + 16, lane, c0, c1);
    const int mb = (lane & 16) ? 8 : 0;
    const int cA = base + (lane & 15);
    const int cB = cA + 16;
#pragma unroll
    for (int r = 0; r < 8; ++r) {
      const int row = mb + r;
      float vA = c0[r] + bo[cA];
      float vB = c1[r] + bo[cB];
      if constexpr (MODE == 1) {
        vA += sRes[row * EMB + cA];
        vB += sRes[row * EMB + cB];
      }
      sT[row * EMB + cA] = vA;
      sT[row * EMB + cB] = vB;
    }
  }
  __syncthreads();

  // ---- 5. LayerNorm 1 -> sH (f32) + sH16 (f16) ----
  if (tid < 16) {
    const float* row = sT + tid * EMB;
    float m = 0.f;
    for (int c = 0; c < EMB; ++c) m += row[c];
    m *= (1.f / EMB);
    float var = 0.f;
    for (int c = 0; c < EMB; ++c) { float d = row[c] - m; var += d * d; }
    var *= (1.f / EMB);
    const float rstd = rsqrtf(var + 1e-5f);
    for (int c = 0; c < EMB; ++c) {
      float hv = (row[c] - m) * rstd * g1[c] + b1v[c];
      sH[tid * EMB + c]   = hv;
      sH16[tid * EMB + c] = (_Float16)hv;
    }
  }
  __syncthreads();

  // ---- 6. FF1: h @ ff1 + bf1, ReLU -> su.f (f16, 16x512) ----
  for (int pair = 0; pair < 4; ++pair) {
    const int n0a = wave * 128 + pair * 32;
    const int n0b = n0a + 16;
    v8f c0 = {}, c1 = {};
    gemm2<EMB>(sH16, EMB, ff1T, EMB, n0a, n0b, lane, c0, c1);
    const int mb = (lane & 16) ? 8 : 0;
    const int cA = n0a + (lane & 15);
    const int cB = cA + 16;
#pragma unroll
    for (int r = 0; r < 8; ++r) {
      const int row = mb + r;
      su.f[row * FF + cA] = (_Float16)fmaxf(c0[r] + bf1[cA], 0.f);
      su.f[row * FF + cB] = (_Float16)fmaxf(c1[r] + bf1[cB], 0.f);
    }
  }
  __syncthreads();

  // ---- 7. FF2: f @ ff2 + bf2 + h -> sT ----
  {
    const int base = wave * 32;
    v8f c0 = {}, c1 = {};
    gemm2<FF>(su.f, FF, ff2T, FF, base, base + 16, lane, c0, c1);
    const int mb = (lane & 16) ? 8 : 0;
    const int cA = base + (lane & 15);
    const int cB = cA + 16;
#pragma unroll
    for (int r = 0; r < 8; ++r) {
      const int row = mb + r;
      sT[row * EMB + cA] = c0[r] + bf2[cA] + sH[row * EMB + cA];
      sT[row * EMB + cB] = c1[r] + bf2[cB] + sH[row * EMB + cB];
    }
  }
  __syncthreads();

  // ---- 8. LayerNorm 2 -> output ----
  if (tid < 16) {
    const float* row = sT + tid * EMB;
    float m = 0.f;
    for (int c = 0; c < EMB; ++c) m += row[c];
    m *= (1.f / EMB);
    float var = 0.f;
    for (int c = 0; c < EMB; ++c) { float d = row[c] - m; var += d * d; }
    var *= (1.f / EMB);
    const float rstd = rsqrtf(var + 1e-5f);
    const size_t base = (size_t)win * (16 * EMB) + tid * EMB;
    for (int c = 0; c < EMB; ++c) {
      float ov = (row[c] - m) * rstd * g2[c] + b2v[c];
      if constexpr (MODE == 0) {
        outF32[base + c] = ov;
      } else {
        outF16[base + c] = (_Float16)(sRes[tid * EMB + c] + ov);  // x1 + block2
      }
    }
  }
}

// ---------------------------------------------------------------------------
// Final projection: y[8192, 2048] @ wf[2048, 128] + bf, write (B, EMB, T)
// ---------------------------------------------------------------------------
__global__ void __launch_bounds__(128)
final_kernel(const _Float16* __restrict__ X2,    // [8192][2048] f16
             const _Float16* __restrict__ wfT,   // [128][2048] f16
             const float* __restrict__ bf,
             float* __restrict__ out) {          // [B][EMB][T]
  const int tid  = threadIdx.x;
  const int wave = tid >> 5;
  const int lane = tid & 31;
  const int m0   = blockIdx.x * 16;   // 512 blocks x 16 rows
  const int n0a  = wave * 32;
  const int n0b  = n0a + 16;
  const _Float16* Abase = X2 + (size_t)m0 * 2048;
  v8f c0 = {}, c1 = {};
  for (int k0 = 0; k0 < 2048; k0 += 32) {
    v16h a  = a_frag(Abase, 2048, k0, lane);
    v16h b0 = b_frag(wfT, 2048, n0a, k0, lane);
    v16h b1 = b_frag(wfT, 2048, n0b, k0, lane);
    c0 = wmma_f16(a, b0, c0);
    c1 = wmma_f16(a, b1, c1);
  }
  const int mb = (lane & 16) ? 8 : 0;
  const int eA = n0a + (lane & 15);
  const int eB = eA + 16;
#pragma unroll
  for (int r = 0; r < 8; ++r) {
    const int row = m0 + mb + r;
    const int b = row >> 7;
    const int t = row & 127;
    out[((size_t)b * EMB + eA) * TT + t] = c0[r] + bf[eA];
    out[((size_t)b * EMB + eB) * TT + t] = c1[r] + bf[eB];
  }
}

// ---------------------------------------------------------------------------
// Host launcher
// ---------------------------------------------------------------------------
extern "C" void kernel_launch(void* const* d_in, const int* in_sizes, int n_in,
                              void* d_out, int out_size, void* d_ws, size_t ws_size,
                              hipStream_t stream) {
  (void)in_sizes; (void)n_in; (void)out_size; (void)ws_size;
  const float* enc    = (const float*)d_in[0];
  const float* off    = (const float*)d_in[1];
  const float* b1_wq  = (const float*)d_in[2];
  const float* b1_wk  = (const float*)d_in[3];
  const float* b1_wv  = (const float*)d_in[4];
  const float* b1_wo  = (const float*)d_in[5];
  const float* b1_bo  = (const float*)d_in[6];
  const float* b1_ff1 = (const float*)d_in[7];
  const float* b1_bf1 = (const float*)d_in[8];
  const float* b1_ff2 = (const float*)d_in[9];
  const float* b1_bf2 = (const float*)d_in[10];
  const float* b1_g1  = (const float*)d_in[11];
  const float* b1_b1  = (const float*)d_in[12];
  const float* b1_g2  = (const float*)d_in[13];
  const float* b1_b2  = (const float*)d_in[14];
  const float* b2_wq  = (const float*)d_in[15];
  const float* b2_wk  = (const float*)d_in[16];
  const float* b2_wv  = (const float*)d_in[17];
  const float* b2_wo  = (const float*)d_in[18];
  const float* b2_bo  = (const float*)d_in[19];
  const float* b2_ff1 = (const float*)d_in[20];
  const float* b2_bf1 = (const float*)d_in[21];
  const float* b2_ff2 = (const float*)d_in[22];
  const float* b2_bf2 = (const float*)d_in[23];
  const float* b2_g1  = (const float*)d_in[24];
  const float* b2_b1  = (const float*)d_in[25];
  const float* b2_g2  = (const float*)d_in[26];
  const float* b2_b2  = (const float*)d_in[27];
  const float* wf     = (const float*)d_in[28];
  const float* bf     = (const float*)d_in[29];
  float* out = (float*)d_out;

  // Workspace carve (bump allocator, 256B aligned); total ~101 MB.
  char* ws = (char*)d_ws;
  size_t cur = 0;
  auto carve = [&](size_t bytes) -> void* {
    void* p = ws + cur;
    cur = (cur + bytes + 255) & ~(size_t)255;
    return p;
  };
  _Float16* xp     = (_Float16*)carve((size_t)NB * TP * CINP * 2);
  _Float16* wqT1   = (_Float16*)carve((size_t)EMB * CINP * 2);
  _Float16* wkT1   = (_Float16*)carve((size_t)EMB * CINP * 2);
  _Float16* wvT1   = (_Float16*)carve((size_t)EMB * CINP * 2);
  _Float16* woT1   = (_Float16*)carve((size_t)EMB * EMB * 2);
  _Float16* ff1T1  = (_Float16*)carve((size_t)FF * EMB * 2);
  _Float16* ff2T1  = (_Float16*)carve((size_t)EMB * FF * 2);
  _Float16* wqT2   = (_Float16*)carve((size_t)EMB * EMB * 2);
  _Float16* wkT2   = (_Float16*)carve((size_t)EMB * EMB * 2);
  _Float16* wvT2   = (_Float16*)carve((size_t)EMB * EMB * 2);
  _Float16* woT2   = (_Float16*)carve((size_t)EMB * EMB * 2);
  _Float16* ff1T2  = (_Float16*)carve((size_t)FF * EMB * 2);
  _Float16* ff2T2  = (_Float16*)carve((size_t)EMB * FF * 2);
  _Float16* wfT    = (_Float16*)carve((size_t)EMB * 2048 * 2);
  float*    x1     = (float*)   carve((size_t)NWIN * 16 * EMB * 4);
  _Float16* x2h    = (_Float16*)carve((size_t)NWIN * 16 * EMB * 2);

  // --- prep activations ---
  {
    int total = NB * TP * CINP;
    prep_x<<<(total + 255) / 256, 256, 0, stream>>>(enc, off, xp);
  }
  // --- prep weights: f32 [K][N] -> f16 [N][Kpad] ---
  auto conv = [&](const float* W, _Float16* Wt, int K, int N, int Kpad) {
    int total = N * Kpad;
    prep_w<<<(total + 255) / 256, 256, 0, stream>>>(W, Wt, K, N, Kpad);
  };
  conv(b1_wq,  wqT1,  CIN,  EMB, CINP);
  conv(b1_wk,  wkT1,  CIN,  EMB, CINP);
  conv(b1_wv,  wvT1,  CIN,  EMB, CINP);
  conv(b1_wo,  woT1,  EMB,  EMB, EMB);
  conv(b1_ff1, ff1T1, EMB,  FF,  EMB);
  conv(b1_ff2, ff2T1, FF,   EMB, FF);
  conv(b2_wq,  wqT2,  EMB,  EMB, EMB);
  conv(b2_wk,  wkT2,  EMB,  EMB, EMB);
  conv(b2_wv,  wvT2,  EMB,  EMB, EMB);
  conv(b2_wo,  woT2,  EMB,  EMB, EMB);
  conv(b2_ff1, ff1T2, EMB,  FF,  EMB);
  conv(b2_ff2, ff2T2, FF,   EMB, FF);
  conv(wf,     wfT,   2048, EMB, 2048);

  // --- block 1: windows from Xpad, no residual, f32 out x1 ---
  block_kernel<CINP, 0><<<NWIN, 128, 0, stream>>>(
      xp, nullptr, wqT1, wkT1, wvT1, woT1, b1_bo, ff1T1, b1_bf1, ff2T1, b1_bf2,
      b1_g1, b1_b1, b1_g2, b1_b2, x1, nullptr);

  // --- block 2: input x1, residual, f16 out x2 = x1 + block(x1) ---
  block_kernel<EMB, 1><<<NWIN, 128, 0, stream>>>(
      nullptr, x1, wqT2, wkT2, wvT2, woT2, b2_bo, ff1T2, b2_bf1, ff2T2, b2_bf2,
      b2_g1, b2_b1, b2_g2, b2_b2, nullptr, x2h);

  // --- final projection + transpose to (B, EMB, T) ---
  final_kernel<<<NWIN / 16, 128, 0, stream>>>(x2h, wfT, bf, out);
}
